// WeaveLayer_61830349193917
// MI455X (gfx1250) — matrix operand order repaired and checked by reading, not compiled, not run on hardware
//
#include <hip/hip_runtime.h>

// ---------------------------------------------------------------------------
// WeaveLayer fused bf16-WMMA implementation for gfx1250 (MI455X).
//
//   A = relu(concat[relu(atom@W_AA+b), segsum(relu(pair@W_PA+b))] @ W_A + b)
//   P = relu(concat[relu(api@W_AP+b)+relu(apj@W_AP+b), relu(pair@W_PP+b)] @ W_P + b)
//
// All GEMMs run on v_wmma_f32_16x16x32_bf16. H=50 is padded to N=64 (4 waves,
// one 16-col tile each). K dims padded to multiples of 32 with zeros.
// ---------------------------------------------------------------------------

typedef __bf16 bf16;
typedef __attribute__((ext_vector_type(16))) __bf16 v16bf;
typedef __attribute__((ext_vector_type(8)))  float  v8f;

#define HDIM   50
#define NPAD   64
#define FA     75
#define FA_PAD 96
#define FP     14
#define FP_PAD 32
#define H_PAD  64   // K padding for H=50

__device__ __forceinline__ bf16 f2bf(float x) { return (bf16)x; }

__device__ __forceinline__ v8f wmma_bf16(v16bf a, v16bf b, v8f c) {
  // D = A(16x32 bf16) x B(32x16 bf16) + C(16x16 f32)
  return __builtin_amdgcn_wmma_f32_16x16x32_bf16(false, a, false, b,
                                                 (short)0, c, false, false);
}

// A-matrix fragment (16x32 bf16) from an LDS tile, rows = matrix rows.
// Lane L holds row L&15; K = half*16 + idx (matches load_b_frag).
__device__ __forceinline__ v16bf load_a_frag(const bf16* lds, int row_stride, int kk) {
  const int lane = threadIdx.x & 31;
  const int row = lane & 15, hf = lane >> 4;
  const bf16* base = lds + row * row_stride + kk + hf * 16;
  v16bf a;
#pragma unroll
  for (int r = 0; r < 4; ++r) {
    a[2 * r]     = base[2 * r];
    a[2 * r + 1] = base[2 * r + 1];
    a[8 + 2 * r]     = base[8 + 2 * r];
    a[8 + 2 * r + 1] = base[8 + 2 * r + 1];
  }
  return a;
}

// B-matrix fragment (32x16 bf16) from padded row-major bf16 weights [Kpad][64].
// Lane L holds column n0 + (L&15); K = half*16 + idx.
__device__ __forceinline__ v16bf load_b_frag(const bf16* __restrict__ Wb, int kk, int n0) {
  const int lane = threadIdx.x & 31;
  const int n = n0 + (lane & 15), hf = lane >> 4;
  const bf16* base = Wb + (size_t)(kk + hf * 16) * NPAD + n;
  v16bf b;
#pragma unroll
  for (int i = 0; i < 16; ++i) b[i] = base[i * NPAD];
  return b;
}

// ---------------------------------------------------------------------------
// Prep kernels
// ---------------------------------------------------------------------------
__global__ void zero_f32(float* __restrict__ p, int n) {
  int i = blockIdx.x * blockDim.x + threadIdx.x;
  if (i < n) p[i] = 0.0f;
}

__global__ void cvt_pad(const float* __restrict__ src, bf16* __restrict__ dst,
                        int K, int N, int Kpad) {
  int i = blockIdx.x * blockDim.x + threadIdx.x;
  if (i < Kpad * NPAD) {
    int k = i / NPAD, n = i % NPAD;
    float v = (k < K && n < N) ? src[k * N + n] : 0.0f;
    dst[i] = f2bf(v);
  }
}

// ---------------------------------------------------------------------------
// Pair-side mega-kernel: per 16-pair tile computes
//   AP_ij, AP_ji, PP, PA (WMMA) -> S = relu(ij)+relu(ji) -> P (WMMA)
//   PA relu'd rows atomically segment-summed into pa_sum.
// ---------------------------------------------------------------------------
__global__ __launch_bounds__(128) void weave_pair_kernel(
    const float* __restrict__ atom_f, const float* __restrict__ pair_f,
    const int* __restrict__ pair_seg, const int* __restrict__ atom_to_pair,
    const bf16* __restrict__ Wb_PA, const bf16* __restrict__ Wb_APt,
    const bf16* __restrict__ Wb_APb, const bf16* __restrict__ Wb_PP,
    const bf16* __restrict__ Wb_Pt, const bf16* __restrict__ Wb_Pb,
    const float* __restrict__ b_PA, const float* __restrict__ b_AP,
    const float* __restrict__ b_PP, const float* __restrict__ b_P,
    float* __restrict__ P_out, float* __restrict__ pa_sum,
    int n_pairs, int n_tiles)
{
  __shared__ bf16 afi[16][FA_PAD];
  __shared__ bf16 afj[16][FA_PAD];
  __shared__ bf16 pfl[16][FP_PAD];
  __shared__ bf16 s_l[16][H_PAD];
  __shared__ bf16 pp_l[16][H_PAD];
  __shared__ int  ai_l[16], aj_l[16], segs[16];

  const int tid  = threadIdx.x;
  const int wave = tid >> 5, lane = tid & 31;
  const int n0   = wave * 16;
  const int n    = n0 + (lane & 15);
  const int hf   = lane >> 4;
  const bool ncol = (n < HDIM);

  // Hold all B fragments in VGPRs across the persistent tile loop.
  v16bf B_APt[3], B_APb[3], B_Pt[2], B_Pb[2], B_PP1, B_PA1;
#pragma unroll
  for (int s = 0; s < 3; ++s) {
    B_APt[s] = load_b_frag(Wb_APt, 32 * s, n0);
    B_APb[s] = load_b_frag(Wb_APb, 32 * s, n0);
  }
#pragma unroll
  for (int s = 0; s < 2; ++s) {
    B_Pt[s] = load_b_frag(Wb_Pt, 32 * s, n0);
    B_Pb[s] = load_b_frag(Wb_Pb, 32 * s, n0);
  }
  B_PP1 = load_b_frag(Wb_PP, 0, n0);
  B_PA1 = load_b_frag(Wb_PA, 0, n0);

  const float bias_ap = ncol ? b_AP[n] : 0.0f;
  const float bias_pp = ncol ? b_PP[n] : 0.0f;
  const float bias_pa = ncol ? b_PA[n] : 0.0f;
  const float bias_p  = ncol ? b_P[n]  : 0.0f;

  for (int tile = blockIdx.x; tile < n_tiles; tile += gridDim.x) {
    // --- stage indices -----------------------------------------------------
    if (tid < 16) {
      int p = tile * 16 + tid;
      bool ok = p < n_pairs;
      ai_l[tid] = ok ? atom_to_pair[2 * p]     : 0;
      aj_l[tid] = ok ? atom_to_pair[2 * p + 1] : 0;
      segs[tid] = ok ? pair_seg[p]             : 0;
    }
    __syncthreads();

    // --- gather atom rows (both endpoints) + pair features into LDS (bf16) -
    for (int idx = tid; idx < 16 * FA_PAD; idx += 128) {
      int m = idx / FA_PAD, k = idx % FA_PAD;
      int p = tile * 16 + m;
      float vi = 0.0f, vj = 0.0f;
      if (p < n_pairs && k < FA) {
        vi = atom_f[(size_t)ai_l[m] * FA + k];
        vj = atom_f[(size_t)aj_l[m] * FA + k];
      }
      afi[m][k] = f2bf(vi);
      afj[m][k] = f2bf(vj);
    }
    for (int idx = tid; idx < 16 * FP_PAD; idx += 128) {
      int m = idx / FP_PAD, k = idx % FP_PAD;
      int p = tile * 16 + m;
      float v = (p < n_pairs && k < FP) ? pair_f[(size_t)p * FP + k] : 0.0f;
      pfl[m][k] = f2bf(v);
    }
    __syncthreads();

    // --- first GEMM stage: AP_ij, AP_ji, PP, PA ---------------------------
    v8f acc_ij, acc_ji, acc_pp, acc_pa;
#pragma unroll
    for (int r = 0; r < 8; ++r) {
      acc_ij[r] = bias_ap; acc_ji[r] = bias_ap;
      acc_pp[r] = bias_pp; acc_pa[r] = bias_pa;
    }
#pragma unroll
    for (int s = 0; s < 3; ++s) {
      v16bf a_i = load_a_frag(&afi[0][0], FA_PAD, 32 * s);
      v16bf a_j = load_a_frag(&afj[0][0], FA_PAD, 32 * s);
      acc_ij = wmma_bf16(a_i, B_APt[s], acc_ij);
      acc_ij = wmma_bf16(a_j, B_APb[s], acc_ij);
      acc_ji = wmma_bf16(a_j, B_APt[s], acc_ji);
      acc_ji = wmma_bf16(a_i, B_APb[s], acc_ji);
    }
    {
      v16bf a_p = load_a_frag(&pfl[0][0], FP_PAD, 0);
      acc_pp = wmma_bf16(a_p, B_PP1, acc_pp);
      acc_pa = wmma_bf16(a_p, B_PA1, acc_pa);
    }

    // --- relu, share S / PP via LDS, segment-sum PA atomically ------------
#pragma unroll
    for (int r = 0; r < 8; ++r) {
      int row = r + 8 * hf;                       // C/D layout: vgpr r -> row
      float sij = fmaxf(acc_ij[r], 0.0f) + fmaxf(acc_ji[r], 0.0f);
      float spp = fmaxf(acc_pp[r], 0.0f);
      s_l[row][n]  = f2bf(ncol ? sij : 0.0f);     // zero the K padding
      pp_l[row][n] = f2bf(ncol ? spp : 0.0f);
      int p = tile * 16 + row;
      if (ncol && p < n_pairs) {
        float pav = fmaxf(acc_pa[r], 0.0f);
        unsafeAtomicAdd(&pa_sum[(size_t)segs[row] * HDIM + n], pav);
      }
    }
    __syncthreads();

    // --- second GEMM stage: P = relu(S@W_Pt + PP@W_Pb + b_P) --------------
    v8f acc_p;
#pragma unroll
    for (int r = 0; r < 8; ++r) acc_p[r] = bias_p;
#pragma unroll
    for (int s = 0; s < 2; ++s) {
      acc_p = wmma_bf16(load_a_frag(&s_l[0][0],  H_PAD, 32 * s), B_Pt[s], acc_p);
      acc_p = wmma_bf16(load_a_frag(&pp_l[0][0], H_PAD, 32 * s), B_Pb[s], acc_p);
    }
#pragma unroll
    for (int r = 0; r < 8; ++r) {
      int row = r + 8 * hf;
      int p = tile * 16 + row;
      if (ncol && p < n_pairs)
        P_out[(size_t)p * HDIM + n] = fmaxf(acc_p[r], 0.0f);
    }
    __syncthreads();   // LDS reused next iteration
  }
}

// ---------------------------------------------------------------------------
// Atom-side kernel: AA = relu(atom@W_AA+b) ; A = relu(AA@W_At + PAs@W_Ab + b)
// ---------------------------------------------------------------------------
__global__ __launch_bounds__(128) void weave_atom_kernel(
    const float* __restrict__ atom_f, const float* __restrict__ pa_sum,
    const bf16* __restrict__ Wb_AA, const bf16* __restrict__ Wb_At,
    const bf16* __restrict__ Wb_Ab,
    const float* __restrict__ b_AA, const float* __restrict__ b_A,
    float* __restrict__ A_out, int n_atoms)
{
  __shared__ bf16 at[16][FA_PAD];
  __shared__ bf16 aa_l[16][H_PAD];
  __shared__ bf16 pas[16][H_PAD];

  const int tid  = threadIdx.x;
  const int wave = tid >> 5, lane = tid & 31;
  const int n0   = wave * 16;
  const int n    = n0 + (lane & 15);
  const int hf   = lane >> 4;
  const bool ncol = (n < HDIM);
  const int tile = blockIdx.x;

  v16bf B_AA[3], B_At[2], B_Ab[2];
#pragma unroll
  for (int s = 0; s < 3; ++s) B_AA[s] = load_b_frag(Wb_AA, 32 * s, n0);
#pragma unroll
  for (int s = 0; s < 2; ++s) {
    B_At[s] = load_b_frag(Wb_At, 32 * s, n0);
    B_Ab[s] = load_b_frag(Wb_Ab, 32 * s, n0);
  }
  const float bias_aa = ncol ? b_AA[n] : 0.0f;
  const float bias_a  = ncol ? b_A[n]  : 0.0f;

  // stage atom rows + PA_sum rows
  for (int idx = tid; idx < 16 * FA_PAD; idx += 128) {
    int m = idx / FA_PAD, k = idx % FA_PAD;
    int a = tile * 16 + m;
    float v = (a < n_atoms && k < FA) ? atom_f[(size_t)a * FA + k] : 0.0f;
    at[m][k] = f2bf(v);
  }
  for (int idx = tid; idx < 16 * H_PAD; idx += 128) {
    int m = idx / H_PAD, k = idx % H_PAD;
    int a = tile * 16 + m;
    float v = (a < n_atoms && k < HDIM) ? pa_sum[(size_t)a * HDIM + k] : 0.0f;
    pas[m][k] = f2bf(v);
  }
  __syncthreads();

  // AA GEMM
  v8f acc_aa;
#pragma unroll
  for (int r = 0; r < 8; ++r) acc_aa[r] = bias_aa;
#pragma unroll
  for (int s = 0; s < 3; ++s)
    acc_aa = wmma_bf16(load_a_frag(&at[0][0], FA_PAD, 32 * s), B_AA[s], acc_aa);
#pragma unroll
  for (int r = 0; r < 8; ++r) {
    int row = r + 8 * hf;
    aa_l[row][n] = f2bf(ncol ? fmaxf(acc_aa[r], 0.0f) : 0.0f);
  }
  __syncthreads();

  // A GEMM
  v8f acc_a;
#pragma unroll
  for (int r = 0; r < 8; ++r) acc_a[r] = bias_a;
#pragma unroll
  for (int s = 0; s < 2; ++s) {
    acc_a = wmma_bf16(load_a_frag(&aa_l[0][0], H_PAD, 32 * s), B_At[s], acc_a);
    acc_a = wmma_bf16(load_a_frag(&pas[0][0],  H_PAD, 32 * s), B_Ab[s], acc_a);
  }
#pragma unroll
  for (int r = 0; r < 8; ++r) {
    int row = r + 8 * hf;
    int a = tile * 16 + row;
    if (ncol && a < n_atoms)
      A_out[(size_t)a * HDIM + n] = fmaxf(acc_a[r], 0.0f);
  }
}

// ---------------------------------------------------------------------------
extern "C" void kernel_launch(void* const* d_in, const int* in_sizes, int n_in,
                              void* d_out, int out_size, void* d_ws, size_t ws_size,
                              hipStream_t stream) {
  const float* atom_f       = (const float*)d_in[0];
  const float* pair_f       = (const float*)d_in[1];
  const int*   pair_seg     = (const int*)d_in[2];
  const int*   atom_to_pair = (const int*)d_in[3];
  const float* W_AA = (const float*)d_in[4];
  const float* b_AA = (const float*)d_in[5];
  const float* W_PA = (const float*)d_in[6];
  const float* b_PA = (const float*)d_in[7];
  const float* W_A  = (const float*)d_in[8];
  const float* b_A  = (const float*)d_in[9];
  const float* W_AP = (const float*)d_in[10];
  const float* b_AP = (const float*)d_in[11];
  const float* W_PP = (const float*)d_in[12];
  const float* b_PP = (const float*)d_in[13];
  const float* W_P  = (const float*)d_in[14];
  const float* b_P  = (const float*)d_in[15];

  const int n_atoms = in_sizes[0] / FA;
  const int n_pairs = in_sizes[1] / FP;

  // workspace: [pa_sum f32 n_atoms*50][padded bf16 weight blocks]
  float* pa_sum = (float*)d_ws;
  bf16* wb = (bf16*)((char*)d_ws + (size_t)n_atoms * HDIM * sizeof(float));
  bf16* Wb_AA  = wb;                    // [96][64]
  bf16* Wb_PA  = Wb_AA  + 96 * NPAD;    // [32][64]
  bf16* Wb_At  = Wb_PA  + 32 * NPAD;    // [64][64]
  bf16* Wb_Ab  = Wb_At  + 64 * NPAD;    // [64][64]
  bf16* Wb_APt = Wb_Ab  + 64 * NPAD;    // [96][64]
  bf16* Wb_APb = Wb_APt + 96 * NPAD;    // [96][64]
  bf16* Wb_PP  = Wb_APb + 96 * NPAD;    // [32][64]
  bf16* Wb_Pt  = Wb_PP  + 32 * NPAD;    // [64][64]
  bf16* Wb_Pb  = Wb_Pt  + 64 * NPAD;    // [64][64]

  float* A_out = (float*)d_out;
  float* P_out = A_out + (size_t)n_atoms * HDIM;

  const int nz = n_atoms * HDIM;
  zero_f32<<<(nz + 255) / 256, 256, 0, stream>>>(pa_sum, nz);
  cvt_pad<<<(96 * NPAD + 255) / 256, 256, 0, stream>>>(W_AA, Wb_AA, FA, HDIM, 96);
  cvt_pad<<<(32 * NPAD + 255) / 256, 256, 0, stream>>>(W_PA, Wb_PA, FP, HDIM, 32);
  cvt_pad<<<(64 * NPAD + 255) / 256, 256, 0, stream>>>(W_A,               Wb_At, HDIM, HDIM, 64);
  cvt_pad<<<(64 * NPAD + 255) / 256, 256, 0, stream>>>(W_A + HDIM * HDIM, Wb_Ab, HDIM, HDIM, 64);
  cvt_pad<<<(96 * NPAD + 255) / 256, 256, 0, stream>>>(W_AP,            Wb_APt, FA, HDIM, 96);
  cvt_pad<<<(96 * NPAD + 255) / 256, 256, 0, stream>>>(W_AP + FA * HDIM, Wb_APb, FA, HDIM, 96);
  cvt_pad<<<(32 * NPAD + 255) / 256, 256, 0, stream>>>(W_PP, Wb_PP, FP, HDIM, 32);
  cvt_pad<<<(64 * NPAD + 255) / 256, 256, 0, stream>>>(W_P,               Wb_Pt, HDIM, HDIM, 64);
  cvt_pad<<<(64 * NPAD + 255) / 256, 256, 0, stream>>>(W_P + HDIM * HDIM, Wb_Pb, HDIM, HDIM, 64);

  const int pair_tiles = (n_pairs + 15) / 16;
  const int pgrid = pair_tiles < 2048 ? pair_tiles : 2048;
  weave_pair_kernel<<<pgrid, 128, 0, stream>>>(
      atom_f, pair_f, pair_seg, atom_to_pair,
      Wb_PA, Wb_APt, Wb_APb, Wb_PP, Wb_Pt, Wb_Pb,
      b_PA, b_AP, b_PP, b_P,
      P_out, pa_sum, n_pairs, pair_tiles);

  const int atom_tiles = (n_atoms + 15) / 16;
  weave_atom_kernel<<<atom_tiles, 128, 0, stream>>>(
      atom_f, pa_sum, Wb_AA, Wb_At, Wb_Ab, b_AA, b_A, A_out, n_atoms);
}
